// DecoderMultiHeadSelfAttention_48120813584496
// MI455X (gfx1250) — matrix operand order repaired and checked by reading, not compiled, not run on hardware
//
#include <hip/hip_runtime.h>

#define NS  512
#define NH  8
#define HD  64
#define KVL 32768
#define TT  (KVL + 1)

typedef __attribute__((ext_vector_type(16))) _Float16 v16h;
typedef __attribute__((ext_vector_type(8)))  _Float16 v8h;
typedef __attribute__((ext_vector_type(8)))  float    v8f;
typedef __attribute__((ext_vector_type(4)))  float    f4;
typedef __attribute__((ext_vector_type(4)))  unsigned int u32x4;
typedef __attribute__((ext_vector_type(8)))  int      i32x8;
typedef __attribute__((ext_vector_type(4)))  int      i32x4;

// ws layout (floats)
#define WS_Q    0     // 512: q * HD^-0.5  (scale^2 folded in once)
#define WS_KNEW 512   // 512
#define WS_VNEW 1024  // 512
#define WS_WV   1536  // 512: attention-output accumulator
#define WS_MAX  2048  // 8
#define WS_SUM  2056  // 8

// k_scores_copy geometry
#define WAVES 4
#define GPW   4                  // 16-row groups per wave
#define ROWS_PER_BLOCK (WAVES * GPW * 16)   // 256
#define LROW  260                // padded LDS row stride in f32 (bank shift 4)

// ---------------------------------------------------------------------------
// TDM descriptor helpers (D# groups per CDNA5 ISA)
// 16-row x 256-col f32 tile, global row stride 512 f32.
//  load : pad_enable=1, pad_interval=256dw(code 7), pad_amount=4dw(code 3)
//         -> LDS row stride 260 f32
//  store: tile_dim0=260, tensor_dim0=256 -> 4 pad dwords/row dropped as OOB
// This toolchain exposes the 6-arg builtin form:
//   (u32x4 g0, i32x8 g1, i32x4, i32x4, i32x8, i32 cpol)
// ---------------------------------------------------------------------------
__device__ __forceinline__ void tdm_k_load(unsigned lds_off, const float* gp)
{
    const unsigned w0 = 0x20000u           // data_size = 2 (4B)
                      | (1u << 20)         // pad_enable
                      | (7u << 22)         // pad_interval: 256 dwords
                      | (3u << 25);        // pad_amount:   4 dwords
    const unsigned long long ga = (unsigned long long)(size_t)gp;
    u32x4 g0 = { 1u, lds_off, (unsigned)ga,
                 ((unsigned)(ga >> 32) & 0x01FFFFFFu) | 0x80000000u }; // type=2
    i32x8 g1 = { (int)w0,
                 (int)(256u << 16),   // tensor_dim0 = 256
                 (int)(16u  << 16),   // tensor_dim1 = 16
                 (int)(256u << 16),   // tile_dim0   = 256
                 16,                  // tile_dim1   = 16
                 512,                 // tensor_dim0_stride = 512
                 0, 0 };
    i32x4 z4 = {0, 0, 0, 0};
    i32x8 z8 = {0, 0, 0, 0, 0, 0, 0, 0};
    __builtin_amdgcn_tensor_load_to_lds(g0, g1, z4, z4, z8, 0);
}

__device__ __forceinline__ void tdm_k_store(unsigned lds_off, float* gp)
{
    const unsigned w0 = 0x20000u;          // data_size = 2 (4B), no pad
    const unsigned long long ga = (unsigned long long)(size_t)gp;
    u32x4 g0 = { 1u, lds_off, (unsigned)ga,
                 ((unsigned)(ga >> 32) & 0x01FFFFFFu) | 0x80000000u }; // type=2
    i32x8 g1 = { (int)w0,
                 (int)(256u << 16),   // tensor_dim0 = 256 (OOB drops pad cols)
                 (int)(16u  << 16),   // tensor_dim1 = 16
                 (int)(260u << 16),   // tile_dim0   = 260 (incl. LDS pad)
                 16,                  // tile_dim1   = 16
                 512,                 // tensor_dim0_stride = 512
                 0, 0 };
    i32x4 z4 = {0, 0, 0, 0};
    i32x8 z8 = {0, 0, 0, 0, 0, 0, 0, 0};
    __builtin_amdgcn_tensor_store_from_lds(g0, g1, z4, z4, z8, 0);
}

// ---------------------------------------------------------------------------
// 1) q/k/v projections (512x512 GEMVs). blocks 0-1: q, 2-3: k, 4-5: v.
// ---------------------------------------------------------------------------
__global__ void k_proj(const float* __restrict__ x,
                       const float* __restrict__ wq, const float* __restrict__ bq,
                       const float* __restrict__ wk,
                       const float* __restrict__ wv, const float* __restrict__ bv,
                       float* __restrict__ ws,
                       float* __restrict__ k_last, float* __restrict__ v_last)
{
    const int which = blockIdx.x >> 1;
    const int row   = ((blockIdx.x & 1) << 8) + threadIdx.x;
    const float* w  = (which == 0) ? wq : (which == 1) ? wk : wv;
    const f4* xv = (const f4*)x;
    const f4* wr = (const f4*)(w + (size_t)row * NS);
    float acc = 0.f;
#pragma unroll 8
    for (int i = 0; i < NS / 4; ++i) {
        f4 a = xv[i], b = wr[i];
        acc += a.x * b.x + a.y * b.y + a.z * b.z + a.w * b.w;
    }
    if (which == 0) {
        ws[WS_Q + row]  = (acc + bq[row]) * 0.125f;  // fold scale^2 = HD^-0.5
        ws[WS_WV + row] = 0.f;                       // reset WV accumulator
    } else if (which == 1) {
        ws[WS_KNEW + row] = acc;
        k_last[row] = acc;                           // k output row 32768
    } else {
        acc += bv[row];
        ws[WS_VNEW + row] = acc;
        v_last[row] = acc;                           // v output row 32768
    }
}

// ---------------------------------------------------------------------------
// 2) score for the appended token (one wave)
// ---------------------------------------------------------------------------
__global__ void k_qk_new(const float* __restrict__ ws, float* __restrict__ qk)
{
    const int h = threadIdx.x;
    if (h < NH) {
        float s = 0.f;
#pragma unroll 8
        for (int d = 0; d < HD; ++d)
            s += ws[WS_Q + h * HD + d] * ws[WS_KNEW + h * HD + d];
        qk[(size_t)h * TT + KVL] = s;
    }
}

// ---------------------------------------------------------------------------
// 3) TDM-pipelined stream: K cache -> LDS (tensor_load), WMMA scores from LDS,
//    LDS -> k output (tensor_store). 4 waves x 4 row-groups x 2 half-tiles,
//    double-buffered; TENSORcnt in-order => wait<=2 steady state and safe WAR.
//    C[t][h] = sum_k Ktile_f16[t][k] * Bq[k][h]; B chunk c nonzero in col c/2.
// ---------------------------------------------------------------------------
__global__ void __launch_bounds__(WAVES * 32)
k_scores_copy(const float* __restrict__ kc, const float* __restrict__ vc,
              float* __restrict__ k_out, float* __restrict__ v_out,
              float* __restrict__ qk, const float* __restrict__ ws)
{
    __shared__ float    kbuf[WAVES][2][16 * LROW];
    __shared__ _Float16 qlds[NS];

    const int tid = threadIdx.x;
#pragma unroll
    for (int i = 0; i < 4; ++i)
        qlds[tid + i * 128] = (_Float16)ws[WS_Q + tid + i * 128];
    __syncthreads();

    const int wave = __builtin_amdgcn_readfirstlane(tid >> 5);
    const int lane = tid & 31;
    const int hi = lane >> 4, lr = lane & 15;

    const size_t waveBase = (size_t)blockIdx.x * ROWS_PER_BLOCK + (size_t)wave * (GPW * 16);
    const unsigned lds0 = (unsigned)(size_t)&kbuf[wave][0][0];
    const unsigned lds1 = (unsigned)(size_t)&kbuf[wave][1][0];

    // prime the pipeline: tiles 0 and 1
    tdm_k_load(lds0, kc + waveBase * NS);
    tdm_k_load(lds1, kc + waveBase * NS + 256);

    for (int g = 0; g < GPW; ++g) {
        const size_t rowBase = waveBase + (size_t)g * 16;
        v8f c = {0.f, 0.f, 0.f, 0.f, 0.f, 0.f, 0.f, 0.f};

#pragma unroll
        for (int half = 0; half < 2; ++half) {
            const int t = g * 2 + half;
            if (t == 0) __builtin_amdgcn_s_wait_tensorcnt(1);
            else        __builtin_amdgcn_s_wait_tensorcnt(2);

            const float* buf = &kbuf[wave][t & 1][0];

#pragma unroll
            for (int ch = 0; ch < 8; ++ch) {
                const float* rp = buf + lr * LROW + ch * 32 + 8 * hi;
                f4 a0 = *(const f4*)(rp);
                f4 a1 = *(const f4*)(rp + 4);
                f4 a2 = *(const f4*)(rp + 16);
                f4 a3 = *(const f4*)(rp + 20);
                v16h a;
                a[0]=(_Float16)a0.x;  a[1]=(_Float16)a0.y;
                a[2]=(_Float16)a0.z;  a[3]=(_Float16)a0.w;
                a[4]=(_Float16)a1.x;  a[5]=(_Float16)a1.y;
                a[6]=(_Float16)a1.z;  a[7]=(_Float16)a1.w;
                a[8]=(_Float16)a2.x;  a[9]=(_Float16)a2.y;
                a[10]=(_Float16)a2.z; a[11]=(_Float16)a2.w;
                a[12]=(_Float16)a3.x; a[13]=(_Float16)a3.y;
                a[14]=(_Float16)a3.z; a[15]=(_Float16)a3.w;

                const int chg = half * 8 + ch;      // global 32-dim chunk 0..15
                v16h b;
#pragma unroll
                for (int e = 0; e < 16; ++e) b[e] = (_Float16)0.f;
                if (lr == (chg >> 1)) {
                    const _Float16* qp = &qlds[(chg >> 1) * HD + (chg & 1) * 32 + hi * 8];
                    v8h q0 = *(const v8h*)qp;
                    v8h q1 = *(const v8h*)(qp + 16);
#pragma unroll
                    for (int e = 0; e < 8; ++e) { b[e] = q0[e]; b[e + 8] = q1[e]; }
                }

                c = __builtin_amdgcn_wmma_f32_16x16x32_f16(false, a, false, b,
                                                           (short)0, c, false, false);
            }

            // write this half-tile back to the k output (reads padded LDS,
            // drops the 4 pad dwords per row as tensor-OOB)
            tdm_k_store((t & 1) ? lds1 : lds0,
                        k_out + rowBase * NS + (size_t)half * 256);
            // refill the buffer with tile t+2 (in-order TDM protects WAR)
            const int tn = t + 2;
            if (tn < 2 * GPW) {
                const size_t rb = waveBase + (size_t)(tn >> 1) * 16;
                tdm_k_load((tn & 1) ? lds1 : lds0,
                           kc + rb * NS + (size_t)(tn & 1) * 256);
            }
        }

        // C layout: VGPR r -> t = r + 8*hi, head = lane%16 (heads 0-7 valid)
        if (lr < NH) {
#pragma unroll
            for (int r = 0; r < 8; ++r)
                qk[(size_t)lr * TT + rowBase + r + 8 * hi] = c[r];
        }
    }

    // block-wide V copy (regular loads keep V hot in the 192MB L2 for k_wv;
    // NT stores keep the 64MB output stream from evicting it)
    const f4* vs = (const f4*)(vc + (size_t)blockIdx.x * ROWS_PER_BLOCK * NS);
    f4*       vd = (f4*)(v_out + (size_t)blockIdx.x * ROWS_PER_BLOCK * NS);
    for (int i = tid; i < ROWS_PER_BLOCK * NS / 4; i += WAVES * 32)
        __builtin_nontemporal_store(vs[i], vd + i);
    // S_ENDPGM performs an implicit wait-idle, covering outstanding TENSORcnt
}

// ---------------------------------------------------------------------------
// 4) per-head softmax stats (max, sum-exp) over 32769 scores
// ---------------------------------------------------------------------------
__global__ void k_softmax_stats(const float* __restrict__ qk, float* __restrict__ ws)
{
    __shared__ float red[256];
    const int h = blockIdx.x, tid = threadIdx.x;
    const float* row = qk + (size_t)h * TT;

    float m = -3.4e38f;
    for (int t = tid; t < TT; t += 256) m = fmaxf(m, row[t]);
    red[tid] = m; __syncthreads();
    for (int s = 128; s > 0; s >>= 1) {
        if (tid < s) red[tid] = fmaxf(red[tid], red[tid + s]);
        __syncthreads();
    }
    m = red[0]; __syncthreads();

    float sum = 0.f;
    for (int t = tid; t < TT; t += 256) sum += __expf(row[t] - m);
    red[tid] = sum; __syncthreads();
    for (int s = 128; s > 0; s >>= 1) {
        if (tid < s) red[tid] += red[tid + s];
        __syncthreads();
    }
    if (tid == 0) { ws[WS_MAX + h] = m; ws[WS_SUM + h] = red[0]; }
}

// ---------------------------------------------------------------------------
// 5) WV: each of 128 threads owns 4 output dims; blocks own 512-row T chunks
// ---------------------------------------------------------------------------
__global__ void k_wv(const float* __restrict__ qk, const float* __restrict__ v,
                     float* __restrict__ ws)
{
    const int tid = threadIdx.x;          // 0..127
    const int h   = tid >> 4;             // (4*tid)/64
    const float m   = ws[WS_MAX + h];
    const float inv = 1.f / ws[WS_SUM + h];
    const float* qrow = qk + (size_t)h * TT;

    const int t0 = blockIdx.x * 512;
    int t1 = t0 + 512; if (t1 > TT) t1 = TT;

    f4 acc = {0.f, 0.f, 0.f, 0.f};
    for (int t = t0; t < t1; ++t) {
        const float w = __expf(qrow[t] - m) * inv;
        f4 vv = *(const f4*)(v + (size_t)t * NS + tid * 4);
        acc.x += w * vv.x; acc.y += w * vv.y; acc.z += w * vv.z; acc.w += w * vv.w;
    }
    atomicAdd(&ws[WS_WV + tid * 4 + 0], acc.x);
    atomicAdd(&ws[WS_WV + tid * 4 + 1], acc.y);
    atomicAdd(&ws[WS_WV + tid * 4 + 2], acc.z);
    atomicAdd(&ws[WS_WV + tid * 4 + 3], acc.w);
}

// ---------------------------------------------------------------------------
// 6) output projection: out = wv_acc @ wo^T + bo
// ---------------------------------------------------------------------------
__global__ void k_out_proj(const float* __restrict__ ws, const float* __restrict__ wo,
                           const float* __restrict__ bo, float* __restrict__ out)
{
    const int i = blockIdx.x * 256 + threadIdx.x;
    const f4* wvv  = (const f4*)(ws + WS_WV);
    const f4* wrow = (const f4*)(wo + (size_t)i * NS);
    float acc = 0.f;
#pragma unroll 8
    for (int j = 0; j < NS / 4; ++j) {
        f4 a = wvv[j], b = wrow[j];
        acc += a.x * b.x + a.y * b.y + a.z * b.z + a.w * b.w;
    }
    out[i] = acc + bo[i];
}

// ---------------------------------------------------------------------------
extern "C" void kernel_launch(void* const* d_in, const int* in_sizes, int n_in,
                              void* d_out, int out_size, void* d_ws, size_t ws_size,
                              hipStream_t stream)
{
    const float* x  = (const float*)d_in[0];
    const float* kc = (const float*)d_in[1];
    const float* vc = (const float*)d_in[2];
    const float* wq = (const float*)d_in[3];
    const float* bq = (const float*)d_in[4];
    const float* wk = (const float*)d_in[5];
    const float* wv = (const float*)d_in[6];
    const float* bv = (const float*)d_in[7];
    const float* wo = (const float*)d_in[8];
    const float* bo = (const float*)d_in[9];

    float* out  = (float*)d_out;                 // [512]
    float* qk   = out + NS;                      // [8, 32769]
    float* kout = qk + (size_t)NH * TT;          // [32769, 512]
    float* vout = kout + (size_t)TT * NS;        // [32769, 512]
    float* ws   = (float*)d_ws;

    k_proj<<<6, 256, 0, stream>>>(x, wq, bq, wk, wv, bv, ws,
                                  kout + (size_t)KVL * NS, vout + (size_t)KVL * NS);
    k_qk_new<<<1, 32, 0, stream>>>(ws, qk);
    k_scores_copy<<<KVL / ROWS_PER_BLOCK, WAVES * 32, 0, stream>>>(kc, vc, kout, vout, qk, ws);
    k_softmax_stats<<<NH, 256, 0, stream>>>(qk, ws);
    k_wv<<<(TT + 511) / 512, 128, 0, stream>>>(qk, vout, ws);
    k_out_proj<<<2, 256, 0, stream>>>(ws, wo, bo, out);
}
